// _DiT_34213709480733
// MI455X (gfx1250) — compile-verified, matrix operand order checked by reading, not run on hardware
//
#include <hip/hip_runtime.h>
#include <hip/hip_bf16.h>

typedef __attribute__((ext_vector_type(16))) __bf16 v16bf;
typedef __attribute__((ext_vector_type(8)))  float  v8f;
typedef __attribute__((ext_vector_type(4))) unsigned int u32x4;
typedef __attribute__((ext_vector_type(4))) int i32x4;
typedef __attribute__((ext_vector_type(8))) int i32x8;

#define SCALE 0.125f   // hd=64 -> 1/sqrt(64)

#if defined(__has_builtin)
#if __has_builtin(__builtin_amdgcn_tensor_load_to_lds) && \
    __has_builtin(__builtin_amdgcn_s_wait_tensorcnt)
#define HAS_TDM 1
#endif
#endif
#ifndef HAS_TDM
#define HAS_TDM 0
#endif

// ---------------------------------------------------------------------------
// WMMA helpers.  Fragment convention (from ISA 16-bit A layout, wave32):
//   lane l holds row m = l&15 (A) / col n = l&15 (B);
//   elements 0..7  = K  (l>=16?8:0) + 0..7
//   elements 8..15 = K  (l>=16?8:0) + 16..23
// -> each fragment is two contiguous 16B loads from a [row][K] layout.
// C/D: lane l, vgpr r -> row = r + (l>=16?8:0), col = l&15.
// ---------------------------------------------------------------------------
__device__ __forceinline__ v8f zero8() {
  v8f z;
#pragma unroll
  for (int i = 0; i < 8; ++i) z[i] = 0.0f;
  return z;
}

__device__ __forceinline__ v16bf frag_ld(const __bf16* base, int stride, int lane) {
  int r   = lane & 15;
  int klo = (lane & 16) ? 8 : 0;
  const __bf16* p = base + (size_t)r * stride + klo;
  union { v16bf v; uint4 q[2]; } u;
  u.q[0] = *(const uint4*)(p);
  u.q[1] = *(const uint4*)(p + 16);
  return u.v;
}

__device__ __forceinline__ v8f wmma_bf16(v16bf a, v16bf b, v8f c) {
  return __builtin_amdgcn_wmma_f32_16x16x32_bf16(false, a, false, b, (short)0, c,
                                                 false, false);
}

// ---------------------------------------------------------------------------
// Tensor Data Mover: async 2-D bf16 tile load (global -> LDS), ISA ch.8 D#.
// group0: [count=1][lds_addr][global_addr 56:0 | type=2]
// group1: data_size=1(2B), tensor_dim0/1, tile_dim0/1, dim0 stride.
// LDS byte address = low 32 bits of the generic pointer (flat aperture rule).
// ---------------------------------------------------------------------------
__device__ __forceinline__ void tdm_load_2d_bf16(void* lds_ptr, const void* gptr,
                                                 unsigned tile_k, unsigned tile_rows,
                                                 unsigned tensor_k, unsigned tensor_rows,
                                                 unsigned row_stride_elems) {
#if HAS_TDM
  unsigned long long ga = (unsigned long long)gptr;
  unsigned lds_addr = (unsigned)(unsigned long long)lds_ptr;
  u32x4 g0;
  g0[0] = 1u;                                             // count=1 (valid user D#)
  g0[1] = lds_addr;                                       // LDS byte address
  g0[2] = (unsigned)(ga & 0xffffffffu);                   // global_addr[31:0]
  g0[3] = (unsigned)((ga >> 32) & 0x01ffffffu) | (2u << 30);  // [56:32] | type=2
  i32x8 g1;
  g1[0] = (int)(1u << 16);                                // data_size=1 -> 2 bytes
  g1[1] = (int)((tensor_k & 0xffffu) << 16);              // tensor_dim0[15:0]
  g1[2] = (int)((tensor_k >> 16) | ((tensor_rows & 0xffffu) << 16)); // dim0 hi|dim1 lo
  g1[3] = (int)((tensor_rows >> 16) | ((tile_k & 0xffffu) << 16));   // dim1 hi|tile0
  g1[4] = (int)(tile_rows & 0xffffu);                     // tile_dim1 (tile_dim2=0)
  g1[5] = (int)row_stride_elems;                          // dim0_stride[31:0]
  g1[6] = 0;                                              // stride0 hi | stride1 lo
  g1[7] = 0;
  i32x4 z4 = {0, 0, 0, 0};
#if __clang_major__ >= 23
  i32x8 z8 = {0, 0, 0, 0, 0, 0, 0, 0};
  __builtin_amdgcn_tensor_load_to_lds(g0, g1, z4, z4, z8, 0);
#else
  __builtin_amdgcn_tensor_load_to_lds(g0, g1, z4, z4, 0);
#endif
#else
  (void)lds_ptr; (void)gptr; (void)tile_k; (void)tile_rows;
  (void)tensor_k; (void)tensor_rows; (void)row_stride_elems;
#endif
}

__device__ __forceinline__ void tdm_wait0() {
#if HAS_TDM
  __builtin_amdgcn_s_wait_tensorcnt(0);
#endif
}

// ---------------------------------------------------------------------------
// Weight prep: transpose to [N][K] bf16 so B-fragments load contiguously.
// ---------------------------------------------------------------------------
__global__ void k_wprep_qkv(const float* __restrict__ Wq, const float* __restrict__ Wkv,
                            __bf16* __restrict__ Wt) {
  int idx = blockIdx.x * blockDim.x + threadIdx.x;     // 1536*512
  if (idx >= 1536 * 512) return;
  int n = idx / 512, k = idx % 512;
  float v = (n < 512) ? Wq[(size_t)k * 512 + n] : Wkv[(size_t)k * 1024 + (n - 512)];
  Wt[(size_t)n * 512 + k] = (__bf16)v;
}

__global__ void k_wprep_conv(const float* __restrict__ W, __bf16* __restrict__ Wt) {
  int idx = blockIdx.x * blockDim.x + threadIdx.x;     // 512*6912
  if (idx >= 512 * 6912) return;
  int nch = idx / 6912, k = idx % 6912;
  int tap = k / 768, c = k % 768;
  Wt[idx] = (__bf16)W[((size_t)nch * 768 + c) * 9 + tap];
}

__global__ void k_wprep_proj(const float* __restrict__ W, __bf16* __restrict__ Wt) {
  int idx = blockIdx.x * blockDim.x + threadIdx.x;     // 512*512
  if (idx >= 512 * 512) return;
  int n = idx / 512, k = idx % 512;
  Wt[(size_t)n * 512 + k] = (__bf16)W[(size_t)k * 512 + n];
}

// ---------------------------------------------------------------------------
// GEMM1: qkv = x[32768,512] @ Wt[1536,512]^T   (A f32->bf16 inline, B via TDM)
// ---------------------------------------------------------------------------
__global__ __launch_bounds__(256) void k_gemm_qkv(const float* __restrict__ X,
                                                  const __bf16* __restrict__ Wt,
                                                  __bf16* __restrict__ out) {
  const int N = 1536, K = 512;
  __shared__ __bf16 As[128][32];
  __shared__ __bf16 Bs[128][32];
  int tid = threadIdx.x, lane = tid & 31, wave = tid >> 5;
  int wm = wave & 3, wn = wave >> 2;
  int m0 = blockIdx.y * 128, n0 = blockIdx.x * 128;
  int lr = tid >> 1, lk = (tid & 1) * 16;
  v8f acc[2][4];
#pragma unroll
  for (int i = 0; i < 2; ++i)
#pragma unroll
    for (int j = 0; j < 4; ++j) acc[i][j] = zero8();

  for (int k0 = 0; k0 < K; k0 += 32) {
    // B tile: async TDM copy (128 rows x 32 bf16, row stride K)
    if (wave == 0)
      tdm_load_2d_bf16(&Bs[0][0], Wt + (size_t)n0 * K + k0, 32, 128, K, N, K);
    // A tile: load f32, convert to bf16, stage to LDS
    const float* asrc = X + (size_t)(m0 + lr) * K + k0 + lk;
    if (k0 + 32 < K) __builtin_prefetch(asrc + 32, 0, 1);
    float4 f0 = *(const float4*)(asrc);
    float4 f1 = *(const float4*)(asrc + 4);
    float4 f2 = *(const float4*)(asrc + 8);
    float4 f3 = *(const float4*)(asrc + 12);
    union { __bf16 h[8]; uint4 q; } pa, pb;
    pa.h[0]=(__bf16)f0.x; pa.h[1]=(__bf16)f0.y; pa.h[2]=(__bf16)f0.z; pa.h[3]=(__bf16)f0.w;
    pa.h[4]=(__bf16)f1.x; pa.h[5]=(__bf16)f1.y; pa.h[6]=(__bf16)f1.z; pa.h[7]=(__bf16)f1.w;
    pb.h[0]=(__bf16)f2.x; pb.h[1]=(__bf16)f2.y; pb.h[2]=(__bf16)f2.z; pb.h[3]=(__bf16)f2.w;
    pb.h[4]=(__bf16)f3.x; pb.h[5]=(__bf16)f3.y; pb.h[6]=(__bf16)f3.z; pb.h[7]=(__bf16)f3.w;
    *(uint4*)&As[lr][lk]     = pa.q;
    *(uint4*)&As[lr][lk + 8] = pb.q;
    if (wave == 0) tdm_wait0();
    __syncthreads();
    v16bf af[2], bf[4];
#pragma unroll
    for (int i = 0; i < 2; ++i) af[i] = frag_ld(&As[wm * 32 + i * 16][0], 32, lane);
#pragma unroll
    for (int j = 0; j < 4; ++j) bf[j] = frag_ld(&Bs[wn * 64 + j * 16][0], 32, lane);
#pragma unroll
    for (int i = 0; i < 2; ++i)
#pragma unroll
      for (int j = 0; j < 4; ++j) acc[i][j] = wmma_bf16(af[i], bf[j], acc[i][j]);
    __syncthreads();
  }
  int ccol = lane & 15, rhi = (lane & 16) ? 8 : 0;
#pragma unroll
  for (int i = 0; i < 2; ++i)
#pragma unroll
    for (int j = 0; j < 4; ++j)
#pragma unroll
      for (int r = 0; r < 8; ++r) {
        int m = m0 + wm * 32 + i * 16 + r + rhi;
        int n = n0 + wn * 64 + j * 16 + ccol;
        out[(size_t)m * N + n] = (__bf16)acc[i][j][r];
      }
}

// ---------------------------------------------------------------------------
// Pool q (from qkv) and context into 49 agents: ain[b][49][768] bf16
// ---------------------------------------------------------------------------
__global__ void k_pool(const __bf16* __restrict__ qkv, const float* __restrict__ ctx,
                       __bf16* __restrict__ ain) {
  int idx = blockIdx.x * blockDim.x + threadIdx.x;
  if (idx >= 32 * 49 * 768) return;
  int c = idx % 768;
  int a = (idx / 768) % 49;
  int b = idx / (768 * 49);
  int p = a / 7, q = a % 7;
  int ys = (p * 32) / 7, ye = ((p + 1) * 32 + 6) / 7;
  int xs = (q * 32) / 7, xe = ((q + 1) * 32 + 6) / 7;
  float s = 0.f;
  for (int y = ys; y < ye; ++y)
    for (int x = xs; x < xe; ++x) {
      int n = y * 32 + x;
      float v = (c < 512) ? (float)qkv[((size_t)(b * 1024 + n)) * 1536 + c]
                          : ctx[((size_t)(b * 1024 + n)) * 256 + (c - 512)];
      s += v;
    }
  s /= (float)((ye - ys) * (xe - xs));
  ain[idx] = (__bf16)s;
}

// ---------------------------------------------------------------------------
// Conv 3x3 over 7x7 agents as 9-tap accumulated GEMM: M=1568, K=768, N=512
// ---------------------------------------------------------------------------
__global__ __launch_bounds__(256) void k_gemm_conv(const __bf16* __restrict__ Ain,
                                                   const __bf16* __restrict__ Wt,
                                                   const float* __restrict__ bias,
                                                   __bf16* __restrict__ agent) {
  const int M = 1568, N = 512;
  __shared__ __bf16 As[128][32];
  __shared__ __bf16 Bs[128][32];
  int tid = threadIdx.x, lane = tid & 31, wave = tid >> 5;
  int wm = wave & 3, wn = wave >> 2;
  int m0 = blockIdx.y * 128, n0 = blockIdx.x * 128;
  int lr = tid >> 1, lk = (tid & 1) * 16;
  v8f acc[2][4];
#pragma unroll
  for (int i = 0; i < 2; ++i)
#pragma unroll
    for (int j = 0; j < 4; ++j) acc[i][j] = zero8();

  for (int tap = 0; tap < 9; ++tap) {
    int dr = tap / 3 - 1, dc = tap % 3 - 1;
    for (int k0 = 0; k0 < 768; k0 += 32) {
      int mg = m0 + lr;
      bool ok = false;
      size_t srcoff = 0;
      if (mg < M) {
        int bb = mg / 49, aa = mg % 49;
        int pp = aa / 7 + dr, qq = aa % 7 + dc;
        if (pp >= 0 && pp < 7 && qq >= 0 && qq < 7) {
          ok = true;
          srcoff = ((size_t)(bb * 49 + pp * 7 + qq)) * 768 + k0 + lk;
        }
      }
      if (ok) {
        *(uint4*)&As[lr][lk]     = *(const uint4*)(Ain + srcoff);
        *(uint4*)&As[lr][lk + 8] = *(const uint4*)(Ain + srcoff + 8);
      } else {
        uint4 z = make_uint4(0, 0, 0, 0);
        *(uint4*)&As[lr][lk] = z;
        *(uint4*)&As[lr][lk + 8] = z;
      }
      const __bf16* bsrc = Wt + (size_t)(n0 + lr) * 6912 + tap * 768 + k0 + lk;
      *(uint4*)&Bs[lr][lk]     = *(const uint4*)(bsrc);
      *(uint4*)&Bs[lr][lk + 8] = *(const uint4*)(bsrc + 8);
      __syncthreads();
      v16bf af[2], bf[4];
#pragma unroll
      for (int i = 0; i < 2; ++i) af[i] = frag_ld(&As[wm * 32 + i * 16][0], 32, lane);
#pragma unroll
      for (int j = 0; j < 4; ++j) bf[j] = frag_ld(&Bs[wn * 64 + j * 16][0], 32, lane);
#pragma unroll
      for (int i = 0; i < 2; ++i)
#pragma unroll
        for (int j = 0; j < 4; ++j) acc[i][j] = wmma_bf16(af[i], bf[j], acc[i][j]);
      __syncthreads();
    }
  }
  int ccol = lane & 15, rhi = (lane & 16) ? 8 : 0;
#pragma unroll
  for (int i = 0; i < 2; ++i)
#pragma unroll
    for (int j = 0; j < 4; ++j)
#pragma unroll
      for (int r = 0; r < 8; ++r) {
        int m = m0 + wm * 32 + i * 16 + r + rhi;
        int n = n0 + wn * 64 + j * 16 + ccol;
        if (m < M) agent[(size_t)m * N + n] = (__bf16)(acc[i][j][r] + bias[n]);
      }
}

// ---------------------------------------------------------------------------
// Positional bias precompute (bilinear 7x7 -> 32x32, half-pixel, edge clamp)
// ---------------------------------------------------------------------------
__device__ __forceinline__ float bilin7(const float* t, int y, int x) {
  float fy = (y + 0.5f) * (7.0f / 32.0f) - 0.5f;
  float fx = (x + 0.5f) * (7.0f / 32.0f) - 0.5f;
  float y0f = floorf(fy), x0f = floorf(fx);
  float wy = fy - y0f, wx = fx - x0f;
  int y0 = (int)y0f, x0 = (int)x0f;
  int y1 = y0 + 1, x1 = x0 + 1;
  y0 = min(max(y0, 0), 6); y1 = min(max(y1, 0), 6);
  x0 = min(max(x0, 0), 6); x1 = min(max(x1, 0), 6);
  float a = t[y0 * 7 + x0], b = t[y0 * 7 + x1];
  float c = t[y1 * 7 + x0], d = t[y1 * 7 + x1];
  return (1.f - wy) * ((1.f - wx) * a + wx * b) + wy * ((1.f - wx) * c + wx * d);
}

__global__ void k_bias_an(const float* __restrict__ an, const float* __restrict__ ah,
                          const float* __restrict__ aw, float* __restrict__ out) {
  int idx = blockIdx.x * blockDim.x + threadIdx.x;   // [8][49][1024]
  if (idx >= 8 * 49 * 1024) return;
  int n = idx % 1024;
  int a = (idx / 1024) % 49;
  int h = idx / (1024 * 49);
  int y = n >> 5, x = n & 31;
  float v = bilin7(an + (size_t)(h * 49 + a) * 49, y, x);
  v += ah[(size_t)(h * 49 + a) * 32 + y];
  v += aw[(size_t)(h * 49 + a) * 32 + x];
  out[idx] = v;
}

__global__ void k_bias_na(const float* __restrict__ na, const float* __restrict__ ha,
                          const float* __restrict__ wa, float* __restrict__ out) {
  int idx = blockIdx.x * blockDim.x + threadIdx.x;   // [8][1024][64] (a padded)
  if (idx >= 8 * 1024 * 64) return;
  int a = idx % 64;
  int n = (idx / 64) % 1024;
  int h = idx / (64 * 1024);
  float v = 0.f;
  if (a < 49) {
    int y = n >> 5, x = n & 31;
    v = bilin7(na + (size_t)(h * 49 + a) * 49, y, x);
    v += ha[(size_t)(h * 32 + y) * 49 + a];
    v += wa[(size_t)(h * 32 + x) * 49 + a];
  }
  out[idx] = v;
}

// ---------------------------------------------------------------------------
// Depthwise 3x3 conv on v (memory bound): dwcb[b][n][512] bf16
// ---------------------------------------------------------------------------
__global__ void k_dwc(const __bf16* __restrict__ qkv, const float* __restrict__ w,
                      const float* __restrict__ bias, __bf16* __restrict__ out) {
  int idx = blockIdx.x * blockDim.x + threadIdx.x;   // 32*1024*512
  if (idx >= 32 * 1024 * 512) return;
  int c = idx & 511;
  int n = (idx >> 9) & 1023;
  int b = idx >> 19;
  int y = n >> 5, x = n & 31;
  float s = bias[c];
#pragma unroll
  for (int t = 0; t < 9; ++t) {
    int yy = y + t / 3 - 1, xx = x + t % 3 - 1;
    if (yy >= 0 && yy < 32 && xx >= 0 && xx < 32)
      s += w[c * 9 + t] *
           (float)qkv[((size_t)(b * 1024 + yy * 32 + xx)) * 1536 + 1024 + c];
  }
  out[idx] = (__bf16)s;
}

// ---------------------------------------------------------------------------
// Agent attention: one workgroup per (b,h).  ~160KB dynamic LDS (CDNA5 WGP
// has 320KB).  All four einsums via WMMA.
// ---------------------------------------------------------------------------
#define ATTN_LDS 163840

__global__ __launch_bounds__(256) void k_attn(const __bf16* __restrict__ qkv,
                                              const __bf16* __restrict__ agent,
                                              const float* __restrict__ bias_an,
                                              const float* __restrict__ bias_na,
                                              const __bf16* __restrict__ dwcb,
                                              __bf16* __restrict__ y) {
  extern __shared__ char smem[];
  __bf16* S   = (__bf16*)smem;               // [64][1024]  scores / probs (stage 1)
  float*  Tld = (float*)smem;                // [128][64]   stage-2 logits (reuse)
  __bf16* P2  = (__bf16*)(smem + 32768);     // [128][64]   stage-2 probs (reuse)
  __bf16* at  = (__bf16*)(smem + 131072);    // [64][64]    agent tokens (a x d)
  __bf16* avt = (__bf16*)(smem + 139264);    // [64][64]    agent_v^T   (d x a)
  __bf16* vt  = (__bf16*)(smem + 147456);    // [64][128]   v^T chunk   (d x n)

  int tid = threadIdx.x, lane = tid & 31, wave = tid >> 5;
  int bh = blockIdx.x, b = bh >> 3, h = bh & 7;
  int ccol = lane & 15, rhi = (lane & 16) ? 8 : 0;

  const __bf16* qbase = qkv + (size_t)b * 1024 * 1536 + h * 64;
  const __bf16* kbase = qbase + 512;
  const __bf16* vbase = qbase + 1024;

  // --- load agent tokens (zero-pad rows 49..63) -------------------------
  {
    int r = tid >> 2, kp = (tid & 3) * 16;
    if (r < 49) {
      const __bf16* src = agent + ((size_t)(b * 49 + r)) * 512 + h * 64 + kp;
      *(uint4*)(at + r * 64 + kp)     = *(const uint4*)(src);
      *(uint4*)(at + r * 64 + kp + 8) = *(const uint4*)(src + 8);
    } else {
      uint4 z = make_uint4(0, 0, 0, 0);
      *(uint4*)(at + r * 64 + kp)     = z;
      *(uint4*)(at + r * 64 + kp + 8) = z;
    }
  }
  __syncthreads();

  // --- Stage 1a: S[a][n] = scale * (a_t @ k^T) + bias_an ----------------
  for (int t = 0; t < 8; ++t) {
    int n0 = (wave * 8 + t) * 16;
    v8f acc[4];
#pragma unroll
    for (int i = 0; i < 4; ++i) acc[i] = zero8();
#pragma unroll
    for (int kk = 0; kk < 64; kk += 32) {
      v16bf bf = frag_ld(kbase + (size_t)n0 * 1536 + kk, 1536, lane);
#pragma unroll
      for (int i = 0; i < 4; ++i) {
        v16bf af = frag_ld(at + i * 16 * 64 + kk, 64, lane);
        acc[i] = wmma_bf16(af, bf, acc[i]);
      }
    }
#pragma unroll
    for (int i = 0; i < 4; ++i)
#pragma unroll
      for (int r = 0; r < 8; ++r) {
        int a = i * 16 + r + rhi;
        int n = n0 + ccol;
        float bv = (a < 49) ? bias_an[((size_t)(h * 49 + a)) * 1024 + n] : 0.f;
        S[a * 1024 + n] = (__bf16)(acc[i][r] * SCALE + bv);
      }
  }
  __syncthreads();

  // --- Stage 1b: softmax over n, per agent row --------------------------
  if (tid < 64) {
    int a = tid;
    float mx = -1e30f;
    for (int n = 0; n < 1024; ++n) {
      float v = (float)S[a * 1024 + n];
      mx = v > mx ? v : mx;
    }
    float sum = 0.f;
    for (int n = 0; n < 1024; ++n) sum += __expf((float)S[a * 1024 + n] - mx);
    float inv = 1.f / sum;
    for (int n = 0; n < 1024; ++n)
      S[a * 1024 + n] = (__bf16)(__expf((float)S[a * 1024 + n] - mx) * inv);
  }

  // --- Stage 1c: agent_v = P @ v (chunked transpose of v through LDS) ---
  int mi = wave & 3, dt0 = (wave >> 2) * 2;
  v8f accav[2];
  accav[0] = zero8(); accav[1] = zero8();
  for (int c = 0; c < 8; ++c) {
    __syncthreads();
    {
      int nloc = tid & 127, dh = (tid >> 7) * 32;
      const __bf16* src = vbase + (size_t)(c * 128 + nloc) * 1536 + dh;
#pragma unroll
      for (int i = 0; i < 32; ++i) vt[(dh + i) * 128 + nloc] = src[i];
    }
    __syncthreads();
#pragma unroll
    for (int kk = 0; kk < 128; kk += 32) {
      v16bf af = frag_ld(S + mi * 16 * 1024 + c * 128 + kk, 1024, lane);
#pragma unroll
      for (int j = 0; j < 2; ++j) {
        v16bf bf = frag_ld(vt + (dt0 + j) * 16 * 128 + kk, 128, lane);
        accav[j] = wmma_bf16(af, bf, accav[j]);
      }
    }
  }
#pragma unroll
  for (int j = 0; j < 2; ++j)
#pragma unroll
    for (int r = 0; r < 8; ++r) {
      int a = mi * 16 + r + rhi;
      int d = (dt0 + j) * 16 + ccol;
      avt[d * 64 + a] = (a < 49) ? (__bf16)accav[j][r] : (__bf16)0.f;
    }

  // --- Stage 2: per n-chunk of 128 rows ---------------------------------
  for (int c = 0; c < 8; ++c) {
    __syncthreads();   // avt/S reads done; T region safe to overwrite
    int n0 = c * 128 + wave * 16;
    v8f tacc[4];
#pragma unroll
    for (int j = 0; j < 4; ++j) tacc[j] = zero8();
#pragma unroll
    for (int kk = 0; kk < 64; kk += 32) {
      v16bf af = frag_ld(qbase + (size_t)n0 * 1536 + kk, 1536, lane);
#pragma unroll
      for (int j = 0; j < 4; ++j) {
        v16bf bf = frag_ld(at + j * 16 * 64 + kk, 64, lane);
        tacc[j] = wmma_bf16(af, bf, tacc[j]);
      }
    }
#pragma unroll
    for (int j = 0; j < 4; ++j)
#pragma unroll
      for (int r = 0; r < 8; ++r) {
        int nl = wave * 16 + r + rhi;
        int a = j * 16 + ccol;
        float bv = bias_na[((size_t)(h * 1024 + c * 128 + nl)) * 64 + a];
        Tld[nl * 64 + a] = tacc[j][r] * SCALE + bv;
      }
    __syncthreads();
    if (tid < 128) {
      int nl = tid;
      float mx = -1e30f;
      for (int a = 0; a < 49; ++a) {
        float v = Tld[nl * 64 + a];
        mx = v > mx ? v : mx;
      }
      float sum = 0.f;
      for (int a = 0; a < 49; ++a) sum += __expf(Tld[nl * 64 + a] - mx);
      float inv = 1.f / sum;
      for (int a = 0; a < 64; ++a)
        P2[nl * 64 + a] =
            (a < 49) ? (__bf16)(__expf(Tld[nl * 64 + a] - mx) * inv) : (__bf16)0.f;
    }
    __syncthreads();
    v8f oacc[4];
#pragma unroll
    for (int j = 0; j < 4; ++j) oacc[j] = zero8();
#pragma unroll
    for (int kk = 0; kk < 64; kk += 32) {
      v16bf af = frag_ld(P2 + wave * 16 * 64 + kk, 64, lane);
#pragma unroll
      for (int j = 0; j < 4; ++j) {
        v16bf bf = frag_ld(avt + j * 16 * 64 + kk, 64, lane);
        oacc[j] = wmma_bf16(af, bf, oacc[j]);
      }
    }
#pragma unroll
    for (int j = 0; j < 4; ++j)
#pragma unroll
      for (int r = 0; r < 8; ++r) {
        int nl = wave * 16 + r + rhi;
        int d = j * 16 + ccol;
        size_t m = (size_t)b * 1024 + c * 128 + nl;
        size_t col = h * 64 + d;
        y[m * 512 + col] = (__bf16)(oacc[j][r] + (float)dwcb[m * 512 + col]);
      }
  }
}

// ---------------------------------------------------------------------------
// Final projection: d_out = y[32768,512] @ proj_w + proj_b   (f32 out)
// Both A and B tiles staged by the Tensor Data Mover.
// ---------------------------------------------------------------------------
__global__ __launch_bounds__(256) void k_gemm_proj(const __bf16* __restrict__ A,
                                                   const __bf16* __restrict__ Wt,
                                                   const float* __restrict__ bias,
                                                   float* __restrict__ out) {
  const int M = 32768, N = 512, K = 512;
  __shared__ __bf16 As[128][32];
  __shared__ __bf16 Bs[128][32];
  int tid = threadIdx.x, lane = tid & 31, wave = tid >> 5;
  int wm = wave & 3, wn = wave >> 2;
  int m0 = blockIdx.y * 128, n0 = blockIdx.x * 128;
  v8f acc[2][4];
#pragma unroll
  for (int i = 0; i < 2; ++i)
#pragma unroll
    for (int j = 0; j < 4; ++j) acc[i][j] = zero8();

  for (int k0 = 0; k0 < K; k0 += 32) {
    if (wave == 0) {
      tdm_load_2d_bf16(&As[0][0], A  + (size_t)m0 * K + k0, 32, 128, K, M, K);
      tdm_load_2d_bf16(&Bs[0][0], Wt + (size_t)n0 * K + k0, 32, 128, K, N, K);
      tdm_wait0();
    }
    __syncthreads();
    v16bf af[2], bf[4];
#pragma unroll
    for (int i = 0; i < 2; ++i) af[i] = frag_ld(&As[wm * 32 + i * 16][0], 32, lane);
#pragma unroll
    for (int j = 0; j < 4; ++j) bf[j] = frag_ld(&Bs[wn * 64 + j * 16][0], 32, lane);
#pragma unroll
    for (int i = 0; i < 2; ++i)
#pragma unroll
      for (int j = 0; j < 4; ++j) acc[i][j] = wmma_bf16(af[i], bf[j], acc[i][j]);
    __syncthreads();
  }
  int ccol = lane & 15, rhi = (lane & 16) ? 8 : 0;
#pragma unroll
  for (int i = 0; i < 2; ++i)
#pragma unroll
    for (int j = 0; j < 4; ++j)
#pragma unroll
      for (int r = 0; r < 8; ++r) {
        int m = m0 + wm * 32 + i * 16 + r + rhi;
        int n = n0 + wn * 64 + j * 16 + ccol;
        out[(size_t)m * N + n] = acc[i][j][r] + bias[n];
      }
}

// ---------------------------------------------------------------------------
// Launch
// ---------------------------------------------------------------------------
extern "C" void kernel_launch(void* const* d_in, const int* in_sizes, int n_in,
                              void* d_out, int out_size, void* d_ws, size_t ws_size,
                              hipStream_t stream) {
  (void)in_sizes; (void)n_in; (void)out_size; (void)ws_size;
  const float* x       = (const float*)d_in[0];
  const float* context = (const float*)d_in[1];
  const float* Wq      = (const float*)d_in[2];
  const float* Wkv     = (const float*)d_in[3];
  const float* conv_w  = (const float*)d_in[4];
  const float* conv_b  = (const float*)d_in[5];
  const float* dwc_w   = (const float*)d_in[6];
  const float* dwc_b   = (const float*)d_in[7];
  const float* proj_w  = (const float*)d_in[8];
  const float* proj_b  = (const float*)d_in[9];
  const float* an_bias = (const float*)d_in[10];
  const float* na_bias = (const float*)d_in[11];
  const float* ah_bias = (const float*)d_in[12];
  const float* aw_bias = (const float*)d_in[13];
  const float* ha_bias = (const float*)d_in[14];
  const float* wa_bias = (const float*)d_in[15];

  unsigned char* ws = (unsigned char*)d_ws;
  __bf16* qkv   = (__bf16*)(ws + 0);            // 32768*1536 bf16
  __bf16* wt1   = (__bf16*)(ws + 100663296);    // 1536*512
  __bf16* ain   = (__bf16*)(ws + 102236160);    // 32*49*768
  __bf16* wt2   = (__bf16*)(ws + 104644608);    // 512*6912
  __bf16* agent = (__bf16*)(ws + 111722496);    // 1568*512
  float*  ban   = (float*) (ws + 113328128);    // 8*49*1024
  float*  bna   = (float*) (ws + 114933760);    // 8*1024*64
  __bf16* dwcb  = (__bf16*)(ws + 117030912);    // 32*1024*512
  __bf16* ybuf  = (__bf16*)(ws + 150585344);    // 32*1024*512
  __bf16* wt3   = (__bf16*)(ws + 184139776);    // 512*512

  k_wprep_qkv<<<3072, 256, 0, stream>>>(Wq, Wkv, wt1);
  k_gemm_qkv<<<dim3(12, 256), 256, 0, stream>>>(x, wt1, qkv);
  k_pool<<<4704, 256, 0, stream>>>(qkv, context, ain);
  k_wprep_conv<<<13824, 256, 0, stream>>>(conv_w, wt2);
  k_gemm_conv<<<dim3(4, 13), 256, 0, stream>>>(ain, wt2, conv_b, agent);
  k_bias_an<<<1568, 256, 0, stream>>>(an_bias, ah_bias, aw_bias, ban);
  k_bias_na<<<2048, 256, 0, stream>>>(na_bias, ha_bias, wa_bias, bna);
  k_dwc<<<65536, 256, 0, stream>>>(qkv, dwc_w, dwc_b, dwcb);
  k_attn<<<256, 256, ATTN_LDS, stream>>>(qkv, agent, ban, bna, dwcb, ybuf);
  k_wprep_proj<<<1024, 256, 0, stream>>>(proj_w, wt3);
  k_gemm_proj<<<dim3(4, 256), 256, 0, stream>>>(ybuf, wt3, proj_b, (float*)d_out);
}
